// RegionAttntion_6012954214875
// MI455X (gfx1250) — compile-verified, matrix operand order checked by reading, not run on hardware
//
#include <hip/hip_runtime.h>
#include <hip/hip_bf16.h>

// ---------------------------------------------------------------------------
// RegionAttention fused pipeline for MI455X (gfx1250, wave32, WMMA bf16)
//   x:(1,36864,512) -> regions (144,256,512) -> QKV -> attn(+EPEG conv,
//   softmax) -> PV -> proj -> un-permute -> (1,36864,512) fp32
// GEMMs: v_wmma_f32_16x16x32_bf16, tiles staged with CDNA5 async-to-LDS DMA
// (global_load_async_to_lds_b128, ASYNCcnt) instead of VGPR round-trips.
// ---------------------------------------------------------------------------

typedef __attribute__((ext_vector_type(16))) __bf16 bf16x16;
typedef __attribute__((ext_vector_type(8)))  float  floatx8;

#define DIMC     512
#define NQKV     1536
#define HEADS    8
#define HD       64
#define RS       16
#define REG_SIDE 12          // 192/16 regions per side
#define NREG     144
#define NTOK     256         // tokens per region
#define NROWS    36864       // total tokens
#define EPEG_K   15
#define SSTRIDE  264         // padded LDS row stride (ushorts), 528B (16B mult)

union FragU  { uint4 u[2]; bf16x16 v; unsigned short s[16]; };
union Acc8U  { floatx8 v; float f[8]; };

__device__ __forceinline__ unsigned short f2bf(float f) {
  unsigned u = __float_as_uint(f);
  unsigned r = ((u >> 16) & 1u) + 0x7FFFu;       // round to nearest even
  return (unsigned short)((u + r) >> 16);
}
__device__ __forceinline__ float bf2f(unsigned short h) {
  return __uint_as_float(((unsigned)h) << 16);
}
__device__ __forceinline__ floatx8 zero8() {
  floatx8 z;
#pragma unroll
  for (int e = 0; e < 8; ++e) z[e] = 0.f;
  return z;
}
__device__ __forceinline__ floatx8 wmma_bf16(bf16x16 a, bf16x16 b, floatx8 c) {
  // D(16x16,f32) = A(16x32,bf16) x B(32x16,bf16) + C
  return __builtin_amdgcn_wmma_f32_16x16x32_bf16(false, a, false, b,
                                                 (short)0, c, false, false);
}
// A/B fragment loader. `base` points at element (tile_origin_row, k0) of a
// row-major matrix with `stride` ushort elements per row. Lane (l&15) owns
// row/col (l&15); K elements interleave {hi*8+e, 16+hi*8+e}: two b128 loads.
__device__ __forceinline__ bf16x16 load_frag(const unsigned short* base, int stride) {
  const int lane = threadIdx.x & 31;
  const unsigned short* p = base + (lane & 15) * stride + ((lane >> 4) << 3);
  FragU f;
  f.u[0] = *reinterpret_cast<const uint4*>(p);
  f.u[1] = *reinterpret_cast<const uint4*>(p + 16);
  return f.v;
}

// CDNA5 async DMA: copy 32 bytes (2 x b128) global -> LDS for this lane.
// IOFFSET is applied to BOTH the LDS and global addresses (ISA 15.18.3),
// so one address pair serves both chunks.  Tracked with ASYNCcnt.
__device__ __forceinline__ void async_copy32(unsigned lds_byte_addr,
                                             const unsigned short* gptr) {
  asm volatile(
      "global_load_async_to_lds_b128 %0, %1, off\n\t"
      "global_load_async_to_lds_b128 %0, %1, off offset:16"
      :
      : "v"(lds_byte_addr), "v"((unsigned long long)gptr)
      : "memory");
}
__device__ __forceinline__ void wait_async0() {
  asm volatile("s_wait_asynccnt 0x0" ::: "memory");
}

// ---------------------------------------------------------------------------
// Kernel 1: region-permute x and convert fp32 -> bf16.  One block per token.
// ---------------------------------------------------------------------------
__global__ __launch_bounds__(256) void pack_x(const float* __restrict__ x,
                                              unsigned short* __restrict__ xr) {
  const int g = blockIdx.x;                 // global token id
  const int i = g / 192, j = g % 192;
  const int r = (i >> 4) * REG_SIDE + (j >> 4);
  const int t = ((i & 15) << 4) | (j & 15);
  const float2 v = reinterpret_cast<const float2*>(x + (size_t)g * DIMC)[threadIdx.x];
  const unsigned packed = ((unsigned)f2bf(v.y) << 16) | (unsigned)f2bf(v.x);
  reinterpret_cast<unsigned*>(xr + ((size_t)(r * NTOK + t)) * DIMC)[threadIdx.x] = packed;
}

__global__ __launch_bounds__(256) void cvt_bf16(const float* __restrict__ in,
                                                unsigned short* __restrict__ out, int n) {
  int i = blockIdx.x * 256 + threadIdx.x;
  if (i < n) out[i] = f2bf(in[i]);
}

// ---------------------------------------------------------------------------
// Kernel 2: QKV GEMM.  C(36864x1536) = Xr(36864x512,bf16) * Wqkv^T + b.
// 128x128 block tile, K-step 32, 8 waves each owning a 32x64 subtile.
// Tiles staged via async-to-LDS DMA; epilogue scatters q (pre-scaled),
// k row-major and v transposed ([rh][d][t]) for the attention kernel.
// ---------------------------------------------------------------------------
__global__ __launch_bounds__(256) void qkv_gemm(
    const unsigned short* __restrict__ X, const unsigned short* __restrict__ W,
    const float* __restrict__ bias,
    unsigned short* __restrict__ Qo, unsigned short* __restrict__ Ko,
    unsigned short* __restrict__ Vt) {
  __shared__ unsigned short At[128][40];    // 32 cols + 8 pad (bank spread)
  __shared__ unsigned short Bt[128][40];
  const int nb = blockIdx.x, mb = blockIdx.y;
  const int tid = threadIdx.x, wave = tid >> 5, lane = tid & 31;
  const int m0 = (wave >> 1) * 32, n0 = (wave & 1) * 64;
  const int ldrow = tid >> 1, ldcol = (tid & 1) * 16;

  floatx8 acc[2][4];
#pragma unroll
  for (int mt = 0; mt < 2; ++mt)
#pragma unroll
    for (int nt = 0; nt < 4; ++nt) acc[mt][nt] = zero8();

  const unsigned short* ga = X + (size_t)(mb * 128 + ldrow) * DIMC + ldcol;
  const unsigned short* gb = W + (size_t)(nb * 128 + ldrow) * DIMC + ldcol;
  // Generic pointers to LDS carry the LDS byte offset in their low 32 bits.
  const unsigned ldsA = (unsigned)(unsigned long long)&At[ldrow][ldcol];
  const unsigned ldsB = (unsigned)(unsigned long long)&Bt[ldrow][ldcol];

#pragma unroll 1
  for (int k0 = 0; k0 < DIMC; k0 += 32) {
    __syncthreads();                        // previous frag reads done
    async_copy32(ldsA, ga + k0);            // X tile -> LDS (DMA, no VGPRs)
    async_copy32(ldsB, gb + k0);            // W tile -> LDS
    if (k0 + 32 < DIMC) {                   // warm L2 for next tiles
      __builtin_prefetch(ga + k0 + 32, 0, 1);
      __builtin_prefetch(gb + k0 + 32, 0, 1);
    }
    wait_async0();
    __syncthreads();
    bf16x16 af[2], bfg[4];
#pragma unroll
    for (int mt = 0; mt < 2; ++mt) af[mt] = load_frag(&At[m0 + mt * 16][0], 40);
#pragma unroll
    for (int nt = 0; nt < 4; ++nt) bfg[nt] = load_frag(&Bt[n0 + nt * 16][0], 40);
#pragma unroll
    for (int mt = 0; mt < 2; ++mt)
#pragma unroll
      for (int nt = 0; nt < 4; ++nt)
        acc[mt][nt] = wmma_bf16(af[mt], bfg[nt], acc[mt][nt]);
  }

  const int hi = lane >> 4, ln = lane & 15;
#pragma unroll
  for (int mt = 0; mt < 2; ++mt)
#pragma unroll
    for (int nt = 0; nt < 4; ++nt) {
      Acc8U a; a.v = acc[mt][nt];
#pragma unroll
      for (int e = 0; e < 8; ++e) {
        const int gm = mb * 128 + m0 + mt * 16 + e + hi * 8;
        const int gn = nb * 128 + n0 + nt * 16 + ln;
        const float val = a.f[e] + bias[gn];
        const int r = gm >> 8, t = gm & 255;
        const int s = gn >> 9, h = (gn >> 6) & 7, d = gn & 63;
        const int rh = r * HEADS + h;
        if (s == 0)       Qo[((size_t)(rh * NTOK + t)) * HD + d] = f2bf(val * 0.125f);
        else if (s == 1)  Ko[((size_t)(rh * NTOK + t)) * HD + d] = f2bf(val);
        else              Vt[((size_t)(rh * HD + d)) * NTOK + t] = f2bf(val);
      }
    }
}

// ---------------------------------------------------------------------------
// Kernel 3: fused attention per (region,head).  S=QK^T in LDS (bf16),
// EPEG depthwise conv fused into an online-softmax scan, O = P*V via WMMA
// with P fragments regenerated on the fly (conv+exp).  137KB dynamic LDS.
// ---------------------------------------------------------------------------
__device__ __forceinline__ float conv_val(const unsigned short* S, int i, int j,
                                          const float* w, float pb) {
  float c = pb + bf2f(S[i * SSTRIDE + j]);   // attn + bias part
#pragma unroll
  for (int t = 0; t < EPEG_K; ++t) {
    const int r2 = i + t - 7;
    if (r2 >= 0 && r2 < NTOK) c = fmaf(w[t], bf2f(S[r2 * SSTRIDE + j]), c);
  }
  return c;
}

__global__ __launch_bounds__(256) void attn_kernel(
    const unsigned short* __restrict__ Q, const unsigned short* __restrict__ K,
    const unsigned short* __restrict__ Vt,
    const float* __restrict__ pe_w, const float* __restrict__ pe_b,
    unsigned short* __restrict__ O) {
  extern __shared__ unsigned char smem[];
  unsigned short* S = reinterpret_cast<unsigned short*>(smem);          // 256x264
  float* rowm = reinterpret_cast<float*>(smem + NTOK * SSTRIDE * 2);    // 256
  float* rsum = rowm + NTOK;                                            // 256

  const int rh = blockIdx.x, h = rh & 7;
  const int tid = threadIdx.x, wave = tid >> 5, lane = tid & 31;
  const int m0 = wave * 32, hi = lane >> 4, ln = lane & 15;
  const unsigned short* Qb = Q + (size_t)rh * NTOK * HD;
  const unsigned short* Kb = K + (size_t)rh * NTOK * HD;
  const unsigned short* Vb = Vt + (size_t)rh * HD * NTOK;

  // --- phase 1: S = (q*scale) K^T, bf16 into LDS --------------------------
  bf16x16 qf[2][2];
#pragma unroll
  for (int mt = 0; mt < 2; ++mt)
#pragma unroll
    for (int kt = 0; kt < 2; ++kt)
      qf[mt][kt] = load_frag(Qb + (m0 + mt * 16) * HD + kt * 32, HD);
#pragma unroll 1
  for (int jt = 0; jt < 16; ++jt) {
    floatx8 a0 = zero8(), a1 = zero8();
#pragma unroll
    for (int kt = 0; kt < 2; ++kt) {
      const bf16x16 bf = load_frag(Kb + (jt * 16) * HD + kt * 32, HD);
      a0 = wmma_bf16(qf[0][kt], bf, a0);
      a1 = wmma_bf16(qf[1][kt], bf, a1);
    }
    Acc8U u0, u1; u0.v = a0; u1.v = a1;
    const int col = jt * 16 + ln;
#pragma unroll
    for (int e = 0; e < 8; ++e) {
      S[(m0 +      e + hi * 8) * SSTRIDE + col] = f2bf(u0.f[e]);
      S[(m0 + 16 + e + hi * 8) * SSTRIDE + col] = f2bf(u1.f[e]);
    }
  }
  __syncthreads();

  // --- phase 2: per-row online softmax stats with fused EPEG conv ---------
  float w[EPEG_K];
#pragma unroll
  for (int t = 0; t < EPEG_K; ++t) w[t] = pe_w[h * EPEG_K + t];
  const float pb = pe_b[h];
  {
    const int i = tid;
    float mx = -3.402823466e38f, sum = 0.f;
#pragma unroll 1
    for (int j = 0; j < NTOK; ++j) {
      const float v = conv_val(S, i, j, w, pb);
      const float nm = fmaxf(mx, v);
      sum = sum * __expf(mx - nm) + __expf(v - nm);
      mx = nm;
    }
    rowm[i] = mx;
    rsum[i] = 1.0f / sum;
  }
  __syncthreads();

  // --- phase 3: O = softmax(S+pe) * V -------------------------------------
  floatx8 acc[2][4];
#pragma unroll
  for (int mt = 0; mt < 2; ++mt)
#pragma unroll
    for (int nt = 0; nt < 4; ++nt) acc[mt][nt] = zero8();

#pragma unroll 1
  for (int ks = 0; ks < 8; ++ks) {
    bf16x16 af[2];
#pragma unroll
    for (int mt = 0; mt < 2; ++mt) {
      const int row = m0 + mt * 16 + ln;
      const float mi = rowm[row], si = rsum[row];
      FragU f;
#pragma unroll
      for (int e = 0; e < 16; ++e) {
        const int j = ks * 32 + (e < 8 ? hi * 8 + e : 16 + hi * 8 + (e - 8));
        f.s[e] = f2bf(__expf(conv_val(S, row, j, w, pb) - mi) * si);
      }
      af[mt] = f.v;
    }
    bf16x16 bfr[4];
#pragma unroll
    for (int nt = 0; nt < 4; ++nt)
      bfr[nt] = load_frag(Vb + (nt * 16) * NTOK + ks * 32, NTOK);
#pragma unroll
    for (int mt = 0; mt < 2; ++mt)
#pragma unroll
      for (int nt = 0; nt < 4; ++nt)
        acc[mt][nt] = wmma_bf16(af[mt], bfr[nt], acc[mt][nt]);
  }

  const int r = rh >> 3;
#pragma unroll
  for (int mt = 0; mt < 2; ++mt)
#pragma unroll
    for (int nt = 0; nt < 4; ++nt) {
      Acc8U a; a.v = acc[mt][nt];
#pragma unroll
      for (int e = 0; e < 8; ++e) {
        const int tkn = m0 + mt * 16 + e + hi * 8;
        const int d = nt * 16 + ln;
        O[((size_t)(r * NTOK + tkn)) * DIMC + h * HD + d] = f2bf(a.f[e]);
      }
    }
}

// ---------------------------------------------------------------------------
// Kernel 4: projection GEMM + bias + inverse region permutation, fp32 out.
// ---------------------------------------------------------------------------
__global__ __launch_bounds__(256) void proj_gemm(
    const unsigned short* __restrict__ A, const unsigned short* __restrict__ W,
    const float* __restrict__ bias, float* __restrict__ out) {
  __shared__ unsigned short At[128][40];
  __shared__ unsigned short Bt[128][40];
  const int nb = blockIdx.x, mb = blockIdx.y;
  const int tid = threadIdx.x, wave = tid >> 5, lane = tid & 31;
  const int m0 = (wave >> 1) * 32, n0 = (wave & 1) * 64;
  const int ldrow = tid >> 1, ldcol = (tid & 1) * 16;

  floatx8 acc[2][4];
#pragma unroll
  for (int mt = 0; mt < 2; ++mt)
#pragma unroll
    for (int nt = 0; nt < 4; ++nt) acc[mt][nt] = zero8();

  const unsigned short* ga = A + (size_t)(mb * 128 + ldrow) * DIMC + ldcol;
  const unsigned short* gb = W + (size_t)(nb * 128 + ldrow) * DIMC + ldcol;
  const unsigned ldsA = (unsigned)(unsigned long long)&At[ldrow][ldcol];
  const unsigned ldsB = (unsigned)(unsigned long long)&Bt[ldrow][ldcol];

#pragma unroll 1
  for (int k0 = 0; k0 < DIMC; k0 += 32) {
    __syncthreads();
    async_copy32(ldsA, ga + k0);
    async_copy32(ldsB, gb + k0);
    if (k0 + 32 < DIMC) {
      __builtin_prefetch(ga + k0 + 32, 0, 1);
      __builtin_prefetch(gb + k0 + 32, 0, 1);
    }
    wait_async0();
    __syncthreads();
    bf16x16 af[2], bfg[4];
#pragma unroll
    for (int mt = 0; mt < 2; ++mt) af[mt] = load_frag(&At[m0 + mt * 16][0], 40);
#pragma unroll
    for (int nt = 0; nt < 4; ++nt) bfg[nt] = load_frag(&Bt[n0 + nt * 16][0], 40);
#pragma unroll
    for (int mt = 0; mt < 2; ++mt)
#pragma unroll
      for (int nt = 0; nt < 4; ++nt)
        acc[mt][nt] = wmma_bf16(af[mt], bfg[nt], acc[mt][nt]);
  }

  const int hi = lane >> 4, ln = lane & 15;
#pragma unroll
  for (int mt = 0; mt < 2; ++mt)
#pragma unroll
    for (int nt = 0; nt < 4; ++nt) {
      Acc8U a; a.v = acc[mt][nt];
#pragma unroll
      for (int e = 0; e < 8; ++e) {
        const int gm = mb * 128 + m0 + mt * 16 + e + hi * 8;
        const int gn = nb * 128 + n0 + nt * 16 + ln;
        const int r = gm >> 8, t = gm & 255;
        const int br = r / REG_SIDE, bc = r % REG_SIDE;
        const int ti = t >> 4, tj = t & 15;
        const int g = (br * 16 + ti) * 192 + (bc * 16 + tj);
        out[(size_t)g * DIMC + gn] = a.f[e] + bias[gn];
      }
    }
}

// ---------------------------------------------------------------------------
// Host launcher.  Workspace layout (bf16 buffers, ~191 MB total):
//   xr 36.0MB | wq 1.5MB | wp 0.5MB | q 36MB | k 36MB | vT 36MB | o 36MB
// ---------------------------------------------------------------------------
extern "C" void kernel_launch(void* const* d_in, const int* in_sizes, int n_in,
                              void* d_out, int out_size, void* d_ws, size_t ws_size,
                              hipStream_t stream) {
  const float* x      = (const float*)d_in[0];
  const float* W_qkv  = (const float*)d_in[1];
  const float* b_qkv  = (const float*)d_in[2];
  const float* W_proj = (const float*)d_in[3];
  const float* b_proj = (const float*)d_in[4];
  const float* pe_w   = (const float*)d_in[5];
  const float* pe_b   = (const float*)d_in[6];
  float* out = (float*)d_out;

  char* p = (char*)d_ws;
  unsigned short* xr = (unsigned short*)p; p += (size_t)NROWS * DIMC * 2;
  unsigned short* wq = (unsigned short*)p; p += (size_t)NQKV * DIMC * 2;
  unsigned short* wp = (unsigned short*)p; p += (size_t)DIMC * DIMC * 2;
  unsigned short* qb = (unsigned short*)p; p += (size_t)NROWS * DIMC * 2;
  unsigned short* kb = (unsigned short*)p; p += (size_t)NROWS * DIMC * 2;
  unsigned short* vt = (unsigned short*)p; p += (size_t)NROWS * DIMC * 2;
  unsigned short* ob = (unsigned short*)p; p += (size_t)NROWS * DIMC * 2;

  pack_x<<<NROWS, 256, 0, stream>>>(x, xr);
  cvt_bf16<<<(NQKV * DIMC + 255) / 256, 256, 0, stream>>>(W_qkv, wq, NQKV * DIMC);
  cvt_bf16<<<(DIMC * DIMC + 255) / 256, 256, 0, stream>>>(W_proj, wp, DIMC * DIMC);

  qkv_gemm<<<dim3(NQKV / 128, NROWS / 128), 256, 0, stream>>>(xr, wq, b_qkv, qb, kb, vt);

  const size_t smem = (size_t)NTOK * SSTRIDE * 2 + 2 * NTOK * sizeof(float); // 137216 B
  (void)hipFuncSetAttribute(reinterpret_cast<const void*>(&attn_kernel),
                            hipFuncAttributeMaxDynamicSharedMemorySize, (int)smem);
  attn_kernel<<<NREG * HEADS, 256, smem, stream>>>(qb, kb, vt, pe_w, pe_b, ob);

  proj_gemm<<<dim3(DIMC / 128, NROWS / 128), 256, 0, stream>>>(ob, wp, b_proj, out);
}
